// FastHybridReasoningLayer_28930899706262
// MI455X (gfx1250) — compile-verified, matrix operand order checked by reading, not run on hardware
//
#include <hip/hip_runtime.h>
#include <hip/hip_bf16.h>
#include <math.h>

// ---------------------------------------------------------------------------
// CDNA5 (gfx1250) fused hybrid-reasoning layer.
// All matmuls run on v_wmma_f32_16x16x32_bf16 (wave32, 16x16 tiles, f32 acc).
// All fragments are loaded as pairs of b128 loads:
//  - A / K^T fragments: contiguous per lane in row-major memory
//  - B fragments: pre-packed into [tile][lane][16] fragment order
// ---------------------------------------------------------------------------

typedef __bf16 bf16_t;
typedef __attribute__((ext_vector_type(16))) __bf16 v16bf;
typedef __attribute__((ext_vector_type(8)))  float  v8f;

static __device__ __forceinline__ v8f wmma_bf16(v16bf a, v16bf b, v8f c) {
  return __builtin_amdgcn_wmma_f32_16x16x32_bf16(false, a, false, b, (short)0, c,
                                                 false, false);
}

static __device__ __forceinline__ v8f v8f_zero() {
  v8f z = {0.f, 0.f, 0.f, 0.f, 0.f, 0.f, 0.f, 0.f};
  return z;
}

union FragU { v16bf v; uint4 u[2]; };

// Fragment element e (0..15) maps to K offset (e<8?0:16) + half*8 + (e&7),
// i.e. two contiguous 8-element (16B) runs at +half*8 and +16+half*8.

// A fragment (16 rows x 32 K), row-major source: lane holds row m0 + lane%16.
static __device__ __forceinline__ v16bf load_a(const bf16_t* A, int lda, int m0,
                                               int k0, int lane) {
  const bf16_t* p =
      A + (size_t)(m0 + (lane & 15)) * lda + k0 + ((lane >> 4) << 3);
  FragU f;
  f.u[0] = *reinterpret_cast<const uint4*>(p);
  f.u[1] = *reinterpret_cast<const uint4*>(p + 16);
  return f.v;
}

// B fragment where logical B[k][n] = Bs[n*ld + k] (keys/rows are contiguous).
static __device__ __forceinline__ v16bf load_b_trans(const bf16_t* Bs, int ld,
                                                     int k0, int n0, int lane) {
  const bf16_t* p =
      Bs + (size_t)(n0 + (lane & 15)) * ld + k0 + ((lane >> 4) << 3);
  FragU f;
  f.u[0] = *reinterpret_cast<const uint4*>(p);
  f.u[1] = *reinterpret_cast<const uint4*>(p + 16);
  return f.v;
}

// B fragment from pre-packed storage:
// packed[(( (n0/16)*(Kdim/32) + k0/32 )*32 + lane)*16 + e]
static __device__ __forceinline__ v16bf load_frag_packed(const bf16_t* P, int Kdim,
                                                         int k0, int n0, int lane) {
  size_t idx =
      (((((size_t)(n0 >> 4) * (size_t)(Kdim >> 5)) + (size_t)(k0 >> 5)) << 5) +
       (size_t)lane)
      << 4;
  FragU f;
  const uint4* p = reinterpret_cast<const uint4*>(P + idx);
  f.u[0] = p[0];
  f.u[1] = p[1];
  return f.v;
}

static __device__ __forceinline__ float redmax16(float v) {
#pragma unroll
  for (int o = 1; o < 16; o <<= 1) v = fmaxf(v, __shfl_xor(v, o, 32));
  return v;
}
static __device__ __forceinline__ float redsum16(float v) {
#pragma unroll
  for (int o = 1; o < 16; o <<= 1) v += __shfl_xor(v, o, 32);
  return v;
}

// ---------------------------------------------------------------------------
// Packing kernels: emit B-operands in WMMA fragment order so the GEMM /
// attention / recurrence inner loops do nothing but b128 loads + WMMA.
// ---------------------------------------------------------------------------
__global__ void pack_b_frag_f32(const float* __restrict__ src, int ld,
                                bf16_t* __restrict__ dst, int K, int N) {
  size_t total = (size_t)K * N;
  size_t i = (size_t)blockIdx.x * blockDim.x + threadIdx.x;
  if (i >= total) return;
  int e = (int)(i & 15);
  int lane = (int)((i >> 4) & 31);
  size_t tile = i >> 9;
  int kt = (int)(tile % (size_t)(K >> 5));
  int nb = (int)(tile / (size_t)(K >> 5));
  int half = lane >> 4;
  int n = (nb << 4) + (lane & 15);
  int k = (kt << 5) + ((e < 8) ? 0 : 16) + (half << 3) + (e & 7);
  dst[i] = (bf16_t)src[(size_t)k * ld + n];
}

// Per-(b,h) pack of a [T x 64] head slice (bf16 source, leading dim ld,
// column base colbase + h*64) into fragment order. dst: [(b*H+h)][T*64].
__global__ void pack_head_bfrag(const bf16_t* __restrict__ src, int ld, int colbase,
                                bf16_t* __restrict__ dst, int T) {
  const int H = 4;
  int h = blockIdx.y, b = blockIdx.z;
  size_t per = (size_t)T * 64;
  const bf16_t* s = src + (size_t)b * T * ld + colbase + h * 64;
  bf16_t* d = dst + (size_t)(b * H + h) * per;
  for (size_t i = (size_t)blockIdx.x * blockDim.x + threadIdx.x; i < per;
       i += (size_t)gridDim.x * blockDim.x) {
    int e = (int)(i & 15);
    int lane = (int)((i >> 4) & 31);
    size_t tile = i >> 9;
    int kt = (int)(tile % (size_t)(T >> 5));
    int nb = (int)(tile / (size_t)(T >> 5));
    int half = lane >> 4;
    int n = (nb << 4) + (lane & 15);
    int k = (kt << 5) + ((e < 8) ? 0 : 16) + (half << 3) + (e & 7);
    d[i] = s[(size_t)k * ld + n];
  }
}

// ---------------------------------------------------------------------------
// Generic WMMA GEMM: C[MxN] = act(A[MxK] @ B[KxN] + bias), B pre-packed.
// 128 threads = 4 waves; block computes a 64x64 C tile (wave = 16 rows x 64 cols).
// act: 0 none, 1 exact GELU, 2 elu(x)+1, 3 sigmoid. Output f32 (Cf) or bf16 (Cb).
// ---------------------------------------------------------------------------
__global__ __launch_bounds__(128)
void gemm_bf16_wmma(const bf16_t* __restrict__ A, int lda,
                    const bf16_t* __restrict__ Bp,
                    const float* __restrict__ bias,
                    float* __restrict__ Cf, bf16_t* __restrict__ Cb, int ldc,
                    int M, int N, int K, int act) {
  int lane = threadIdx.x & 31;
  int wv   = threadIdx.x >> 5;
  int half = lane >> 4;
  int n0 = blockIdx.x * 64;
  int m0 = blockIdx.y * 64 + wv * 16;
  (void)M; (void)N;

  v8f acc[4];
#pragma unroll
  for (int t = 0; t < 4; ++t) acc[t] = v8f_zero();

  for (int k0 = 0; k0 < K; k0 += 32) {
    v16bf a = load_a(A, lda, m0, k0, lane);
    if (k0 + 32 < K)
      __builtin_prefetch(&A[(size_t)(m0 + (lane & 15)) * lda + k0 + 32], 0, 1);
#pragma unroll
    for (int t = 0; t < 4; ++t) {
      v16bf b = load_frag_packed(Bp, K, k0, n0 + 16 * t, lane);
      acc[t] = wmma_bf16(a, b, acc[t]);
    }
  }

#pragma unroll
  for (int t = 0; t < 4; ++t) {
    int n = n0 + 16 * t + (lane & 15);
    float bv = bias ? bias[n] : 0.f;
#pragma unroll
    for (int r = 0; r < 8; ++r) {
      int m = m0 + r + (half << 3);
      float v = acc[t][r] + bv;
      if (act == 1)      v = 0.5f * v * (1.f + erff(v * 0.70710678118654752f));
      else if (act == 2) v = (v > 0.f) ? (v + 1.f) : __expf(v);   // elu(v)+1
      else if (act == 3) v = 1.f / (1.f + __expf(-v));            // sigmoid
      size_t idx = (size_t)m * ldc + n;
      if (Cb) Cb[idx] = (bf16_t)v; else Cf[idx] = v;
    }
  }
}

// ---------------------------------------------------------------------------
// Causal flash attention. qkv: bf16 [B, T, 768] (q|k|v, head h at col h*64).
// V is consumed from per-head packed fragments (vpk). Block = (64 queries,
// head, batch); 4 waves, wave = 16 query rows. Online softmax, 32-key chunks;
// P transposed C->A layout via per-wave LDS. Output: bf16 [B, T, 256].
// ---------------------------------------------------------------------------
__global__ __launch_bounds__(128)
void attn_causal_wmma(const bf16_t* __restrict__ qkv,
                      const bf16_t* __restrict__ vpk,
                      bf16_t* __restrict__ out, int T) {
  __shared__ alignas(16) bf16_t Pls[4][16 * 32];
  int lane = threadIdx.x & 31;
  int wv   = threadIdx.x >> 5;
  int half = lane >> 4;
  int h = blockIdx.y, b = blockIdx.z;
  int qblk = blockIdx.x * 64;
  int q0 = qblk + wv * 16;

  const bf16_t* qb = qkv + (size_t)b * T * 768 + (size_t)h * 64;
  const bf16_t* kb = qb + 256;
  const bf16_t* vh = vpk + (size_t)(b * 4 + h) * T * 64;

  v16bf qa0 = load_a(qb, 768, q0, 0, lane);
  v16bf qa1 = load_a(qb, 768, q0, 32, lane);

  v8f o[4];
  float rmax[8], rsum[8];
#pragma unroll
  for (int t = 0; t < 4; ++t) o[t] = v8f_zero();
#pragma unroll
  for (int r = 0; r < 8; ++r) { rmax[r] = -1e30f; rsum[r] = 0.f; }

  int nch = (qblk + 64) / 32;  // uniform over block; masking handles causality
  for (int j = 0; j < nch; ++j) {
    int key0 = j * 32;
    v8f st[2];
#pragma unroll
    for (int kt = 0; kt < 2; ++kt) {
      int kk = key0 + kt * 16;
      v16bf b0 = load_b_trans(kb, 768, 0,  kk, lane);  // K^T, features 0..31
      v16bf b1 = load_b_trans(kb, 768, 32, kk, lane);  // features 32..63
      v8f s = wmma_bf16(qa0, b0, wmma_bf16(qa1, b1, v8f_zero()));
      int kidx = kk + (lane & 15);
#pragma unroll
      for (int r = 0; r < 8; ++r) {
        float v = s[r] * 0.125f;                       // 1/sqrt(64)
        int qrow = q0 + r + (half << 3);
        if (kidx > qrow) v = -1e30f;
        s[r] = v;
      }
      st[kt] = s;
    }
    // online softmax update + stage P (bf16) for the PV WMMA
#pragma unroll
    for (int r = 0; r < 8; ++r) {
      float cm = redmax16(fmaxf(st[0][r], st[1][r]));
      float nm = fmaxf(rmax[r], cm);
      float ef = __expf(rmax[r] - nm);
      rmax[r] = nm;
      float p0 = __expf(st[0][r] - nm);
      float p1 = __expf(st[1][r] - nm);
      rsum[r] = rsum[r] * ef + redsum16(p0 + p1);
#pragma unroll
      for (int t = 0; t < 4; ++t) o[t][r] *= ef;
      int row = r + (half << 3);
      Pls[wv][row * 32 + (lane & 15)]      = (bf16_t)p0;
      Pls[wv][row * 32 + 16 + (lane & 15)] = (bf16_t)p1;
    }
    __syncthreads();
    v16bf pa = load_a(&Pls[wv][0], 32, 0, 0, lane);    // P as A fragment
#pragma unroll
    for (int t = 0; t < 4; ++t) {
      v16bf vv = load_frag_packed(vh, T, key0, 16 * t, lane);
      o[t] = wmma_bf16(pa, vv, o[t]);
    }
    __syncthreads();
  }

#pragma unroll
  for (int t = 0; t < 4; ++t) {
#pragma unroll
    for (int r = 0; r < 8; ++r) {
      int qrow = q0 + r + (half << 3);
      float v = o[t][r] / rsum[r];
      out[((size_t)b * T + qrow) * 256 + h * 64 + 16 * t + (lane & 15)] = (bf16_t)v;
    }
  }
}

// ---------------------------------------------------------------------------
// Chunked linear recurrence (positive-feature linear attention).
// One block per (b,h); 64-token chunks, sequential over T/64 chunks.
// S kept in LDS: f32 accumulator (row-major) + bf16 operand (fragment-packed).
// out_t = Q_t@S_prev + tril(Q K^T)@V ; denom_t = Q_t.Z_prev + rowsum(tril(QK^T)).
// Q/K feature-mapped (elu+1) bf16; K/V also provided fragment-packed per head.
// ---------------------------------------------------------------------------
__global__ __launch_bounds__(128)
void linrec_wmma(const bf16_t* __restrict__ Qm, const bf16_t* __restrict__ Km,
                 const bf16_t* __restrict__ Kpk, const bf16_t* __restrict__ Vpk,
                 bf16_t* __restrict__ out, int T) {
  __shared__ alignas(16) float  Sf[64 * 64];
  __shared__ alignas(16) bf16_t Sbp[64 * 64];   // fragment-packed (Kdim = 64)
  __shared__ float Zf[64];
  __shared__ float QZ[64];
  __shared__ alignas(16) bf16_t Pst[4][16 * 64];

  int lane = threadIdx.x & 31;
  int wv   = threadIdx.x >> 5;
  int half = lane >> 4;
  int tid  = threadIdx.x;
  int b = blockIdx.x >> 2;      // H = 4
  int h = blockIdx.x & 3;

  for (int i = tid; i < 64 * 64; i += 128) { Sf[i] = 0.f; Sbp[i] = (bf16_t)0.f; }
  if (tid < 64) Zf[tid] = 0.f;
  __syncthreads();

  const bf16_t* qp  = Qm  + (size_t)b * T * 256 + (size_t)h * 64;
  const bf16_t* kp  = Km  + (size_t)b * T * 256 + (size_t)h * 64;
  const bf16_t* kph = Kpk + (size_t)(b * 4 + h) * T * 64;
  const bf16_t* vph = Vpk + (size_t)(b * 4 + h) * T * 64;

  int nchunk = T / 64;
  for (int c = 0; c < nchunk; ++c) {
    int c0 = c * 64;
    const bf16_t* kc = kp + (size_t)c0 * 256;

    // denominator carry: Q_row . Z_prev  (threads 0..63, one row each)
    if (tid < 64) {
      const uint4* qr =
          reinterpret_cast<const uint4*>(qp + (size_t)(c0 + tid) * 256);
      float s = 0.f;
#pragma unroll
      for (int ch = 0; ch < 8; ++ch) {
        union { uint4 u; bf16_t e[8]; } w;
        w.u = qr[ch];
#pragma unroll
        for (int j = 0; j < 8; ++j) s += (float)w.e[j] * Zf[ch * 8 + j];
      }
      QZ[tid] = s;
    }

    int r0 = c0 + wv * 16;
    v16bf qa0 = load_a(qp, 256, r0, 0, lane);
    v16bf qa1 = load_a(qp, 256, r0, 32, lane);
    v8f o[4];
#pragma unroll
    for (int t = 0; t < 4; ++t) o[t] = v8f_zero();

    // o = Q @ S_prev   (S operand is fragment-packed in LDS)
#pragma unroll
    for (int t = 0; t < 4; ++t) {
      v16bf sb0 = load_frag_packed(Sbp, 64, 0,  16 * t, lane);
      v16bf sb1 = load_frag_packed(Sbp, 64, 32, 16 * t, lane);
      o[t] = wmma_bf16(qa1, sb1, wmma_bf16(qa0, sb0, o[t]));
    }

    // A_intra = tril(Q K^T), rowsums, stage to LDS for the P@V WMMA
    float rs[8] = {0.f, 0.f, 0.f, 0.f, 0.f, 0.f, 0.f, 0.f};
#pragma unroll
    for (int kt = 0; kt < 4; ++kt) {
      int kk = kt * 16;
      v16bf b0 = load_b_trans(kc, 256, 0,  kk, lane);
      v16bf b1 = load_b_trans(kc, 256, 32, kk, lane);
      v8f s = wmma_bf16(qa0, b0, wmma_bf16(qa1, b1, v8f_zero()));
      int kl = kk + (lane & 15);
#pragma unroll
      for (int r = 0; r < 8; ++r) {
        int rl = wv * 16 + r + (half << 3);
        float v = (kl <= rl) ? s[r] : 0.f;            // inclusive causal, zero-mask
        rs[r] += v;
        Pst[wv][(r + (half << 3)) * 64 + kl] = (bf16_t)v;
      }
    }
#pragma unroll
    for (int r = 0; r < 8; ++r) rs[r] = redsum16(rs[r]);
    __syncthreads();   // Pst + QZ visible

    // o += A_intra @ V
#pragma unroll
    for (int kk = 0; kk < 64; kk += 32) {
      v16bf pa = load_a(&Pst[wv][0], 64, 0, kk, lane);
#pragma unroll
      for (int t = 0; t < 4; ++t) {
        v16bf vv = load_frag_packed(vph, T, c0 + kk, 16 * t, lane);
        o[t] = wmma_bf16(pa, vv, o[t]);
      }
    }

    // normalize and store
#pragma unroll
    for (int t = 0; t < 4; ++t) {
#pragma unroll
      for (int r = 0; r < 8; ++r) {
        int rl = wv * 16 + r + (half << 3);
        float denom = QZ[rl] + rs[r] + 1e-6f;
        float v = o[t][r] / denom;
        out[((size_t)b * T + c0 + rl) * 256 + h * 64 + 16 * t + (lane & 15)] =
            (bf16_t)v;
      }
    }
    __syncthreads();   // all reads of Sbp/Zf done before state update

    // S += K^T @ V  (wave wv owns feature rows d0..d0+15).
    // K^T as A-operand comes from the packed K (A- and B-fragment lane
    // mappings are identical), so these are b128 loads too.
    {
      int d0 = wv * 16;
      v16bf ka0 = load_frag_packed(kph, T, c0,      d0, lane);
      v16bf ka1 = load_frag_packed(kph, T, c0 + 32, d0, lane);
#pragma unroll
      for (int t = 0; t < 4; ++t) {
        v8f cacc;
#pragma unroll
        for (int r = 0; r < 8; ++r)
          cacc[r] = Sf[(d0 + r + (half << 3)) * 64 + 16 * t + (lane & 15)];
        v16bf v0 = load_frag_packed(vph, T, c0,      16 * t, lane);
        v16bf v1 = load_frag_packed(vph, T, c0 + 32, 16 * t, lane);
        cacc = wmma_bf16(ka1, v1, wmma_bf16(ka0, v0, cacc));
#pragma unroll
        for (int r = 0; r < 8; ++r) {
          int m = d0 + r + (half << 3);               // K-dim index of S
          int n = 16 * t + (lane & 15);               // N-dim index of S
          Sf[m * 64 + n] = cacc[r];
          // scatter into fragment-packed layout for next chunk's Q@S
          int kq = m & 31;
          int e = ((kq >= 16) ? 8 : 0) + (kq & 7);
          int lane2 = (n & 15) + (((kq >> 3) & 1) << 4);
          Sbp[(((n >> 4) * 2 + (m >> 5)) * 32 + lane2) * 16 + e] = (bf16_t)cacc[r];
        }
      }
    }
    // Z += colsum(K)
    if (tid < 64) {
      float z = Zf[tid];
      for (int tk = 0; tk < 64; ++tk) z += (float)kc[(size_t)tk * 256 + tid];
      Zf[tid] = z;
    }
    __syncthreads();
  }
}

// ---------------------------------------------------------------------------
// Elementwise / LayerNorm glue
// ---------------------------------------------------------------------------
__global__ __launch_bounds__(256)
void add_layernorm(const float* __restrict__ x, const float* __restrict__ r,
                   const float* __restrict__ g, const float* __restrict__ bta,
                   float* __restrict__ outf, bf16_t* __restrict__ outb,
                   int ldb, int coloff) {
  __shared__ float red[256];
  int row = blockIdx.x, tid = threadIdx.x;
  float v = x[(size_t)row * 256 + tid] + (r ? r[(size_t)row * 256 + tid] : 0.f);
  red[tid] = v;
  __syncthreads();
#pragma unroll
  for (int s = 128; s > 0; s >>= 1) {
    if (tid < s) red[tid] += red[tid + s];
    __syncthreads();
  }
  float mean = red[0] * (1.f / 256.f);
  __syncthreads();
  float d = v - mean;
  red[tid] = d * d;
  __syncthreads();
#pragma unroll
  for (int s = 128; s > 0; s >>= 1) {
    if (tid < s) red[tid] += red[tid + s];
    __syncthreads();
  }
  float var = red[0] * (1.f / 256.f);
  float y = d * rsqrtf(var + 1e-5f) * g[tid] + bta[tid];
  if (outf) outf[(size_t)row * 256 + tid] = y;
  if (outb) outb[(size_t)row * ldb + coloff + tid] = (bf16_t)y;
}

__global__ void gate_merge(const float* __restrict__ g, const float* __restrict__ p,
                           const float* __restrict__ s, float* __restrict__ mf,
                           bf16_t* __restrict__ mb, size_t n) {
  size_t i = (size_t)blockIdx.x * blockDim.x + threadIdx.x;
  if (i < n) {
    float gv = g[i];
    float v = gv * p[i] + (1.f - gv) * s[i];
    mf[i] = v;
    mb[i] = (bf16_t)v;
  }
}

__global__ void f32_to_bf16(const float* __restrict__ src, bf16_t* __restrict__ dst,
                            size_t n) {
  size_t i = (size_t)blockIdx.x * blockDim.x + threadIdx.x;
  if (i < n) dst[i] = (bf16_t)src[i];
}

// ---------------------------------------------------------------------------
// Host-side orchestration
// ---------------------------------------------------------------------------
extern "C" void kernel_launch(void* const* d_in, const int* in_sizes, int n_in,
                              void* d_out, int out_size, void* d_ws, size_t ws_size,
                              hipStream_t stream) {
  (void)in_sizes; (void)n_in; (void)out_size; (void)ws_size;
  const int Bb = 4, T = 2048, D = 256, H = 4, HID = 1024;
  const int R = Bb * T;  // 8192 token rows

  const float* x     = (const float*)d_in[0];
  const float* qkv_w = (const float*)d_in[1];
  const float* qkv_b = (const float*)d_in[2];
  const float* o_w   = (const float*)d_in[3];
  const float* o_b   = (const float*)d_in[4];
  const float* w1    = (const float*)d_in[5];
  const float* b1    = (const float*)d_in[6];
  const float* w2    = (const float*)d_in[7];
  const float* b2    = (const float*)d_in[8];
  const float* rq_w  = (const float*)d_in[9];
  const float* rq_b  = (const float*)d_in[10];
  const float* rk_w  = (const float*)d_in[11];
  const float* rk_b  = (const float*)d_in[12];
  const float* rv_w  = (const float*)d_in[13];
  const float* rv_b  = (const float*)d_in[14];
  const float* ro_w  = (const float*)d_in[15];
  const float* ro_b  = (const float*)d_in[16];
  const float* cg_w  = (const float*)d_in[17];
  const float* cg_b  = (const float*)d_in[18];
  const float* ln1g  = (const float*)d_in[19];
  const float* ln1b  = (const float*)d_in[20];
  const float* ln2g  = (const float*)d_in[21];
  const float* ln2b  = (const float*)d_in[22];
  const float* ln3g  = (const float*)d_in[23];
  const float* ln3b  = (const float*)d_in[24];

  char* ws = (char*)d_ws;
  size_t off = 0;
  auto alloc = [&](size_t bytes) -> void* {
    void* p = (void*)(ws + off);
    off += (bytes + 255) & ~(size_t)255;
    return p;
  };

  // fragment-packed bf16 weights
  bf16_t* wqkv = (bf16_t*)alloc((size_t)D * 3 * D * 2);
  bf16_t* wo   = (bf16_t*)alloc((size_t)D * D * 2);
  bf16_t* w1b  = (bf16_t*)alloc((size_t)D * HID * 2);
  bf16_t* w2b  = (bf16_t*)alloc((size_t)HID * D * 2);
  bf16_t* wrq  = (bf16_t*)alloc((size_t)D * D * 2);
  bf16_t* wrk  = (bf16_t*)alloc((size_t)D * D * 2);
  bf16_t* wrv  = (bf16_t*)alloc((size_t)D * D * 2);
  bf16_t* wro  = (bf16_t*)alloc((size_t)D * D * 2);
  bf16_t* wcg  = (bf16_t*)alloc((size_t)2 * D * D * 2);
  // activations
  bf16_t* xb      = (bf16_t*)alloc((size_t)R * D * 2);
  bf16_t* qkvb    = (bf16_t*)alloc((size_t)R * 768 * 2);
  bf16_t* vatt    = (bf16_t*)alloc((size_t)R * 64 * H * 2 / H * H);  // [B*H][T*64]
  bf16_t* attnb   = (bf16_t*)alloc((size_t)R * D * 2);
  float*  oproj   = (float*) alloc((size_t)R * D * 4);
  float*  parf    = (float*) alloc((size_t)R * D * 4);
  bf16_t* catb    = (bf16_t*)alloc((size_t)R * 512 * 2);  // [parallel | seq_repr]
  bf16_t* qm      = (bf16_t*)alloc((size_t)R * D * 2);
  bf16_t* km      = (bf16_t*)alloc((size_t)R * D * 2);
  bf16_t* vm      = (bf16_t*)alloc((size_t)R * D * 2);
  bf16_t* kpk     = (bf16_t*)alloc((size_t)R * D * 2);    // [B*H][T*64] packed
  bf16_t* vpk     = (bf16_t*)alloc((size_t)R * D * 2);    // [B*H][T*64] packed
  bf16_t* seqb    = (bf16_t*)alloc((size_t)R * D * 2);
  float*  rop     = (float*) alloc((size_t)R * D * 4);
  float*  seqr    = (float*) alloc((size_t)R * D * 4);
  float*  gatef   = (float*) alloc((size_t)R * D * 4);
  float*  mergedf = (float*) alloc((size_t)R * D * 4);
  bf16_t* mergedb = (bf16_t*)alloc((size_t)R * D * 2);
  bf16_t* ffnh    = (bf16_t*)alloc((size_t)R * HID * 2);
  float*  ffn2    = (float*) alloc((size_t)R * D * 4);

  auto packW = [&](const float* s, bf16_t* dstp, int K, int N) {
    size_t total = (size_t)K * N;
    pack_b_frag_f32<<<(unsigned)((total + 255) / 256), 256, 0, stream>>>(
        s, N, dstp, K, N);
  };
  packW(qkv_w, wqkv, D, 768);
  packW(o_w,   wo,   D, D);
  packW(w1,    w1b,  D, HID);
  packW(w2,    w2b,  HID, D);
  packW(rq_w,  wrq,  D, D);
  packW(rk_w,  wrk,  D, D);
  packW(rv_w,  wrv,  D, D);
  packW(ro_w,  wro,  D, D);
  packW(cg_w,  wcg,  2 * D, D);
  f32_to_bf16<<<(unsigned)(((size_t)R * D + 255) / 256), 256, 0, stream>>>(
      x, xb, (size_t)R * D);

  dim3 blk(128);
  // qkv projection -> bf16 [R,768]
  gemm_bf16_wmma<<<dim3(768 / 64, R / 64), blk, 0, stream>>>(
      xb, D, wqkv, qkv_b, nullptr, qkvb, 768, R, 768, D, 0);
  // pack V heads into fragment order for the PV WMMAs
  pack_head_bfrag<<<dim3(256, H, Bb), 256, 0, stream>>>(qkvb, 768, 512, vatt, T);
  // causal attention -> bf16 [R,256]
  attn_causal_wmma<<<dim3(T / 64, H, Bb), blk, 0, stream>>>(qkvb, vatt, attnb, T);
  // output projection -> f32
  gemm_bf16_wmma<<<dim3(D / 64, R / 64), blk, 0, stream>>>(
      attnb, D, wo, o_b, oproj, nullptr, D, R, D, D, 0);
  // parallel_repr = LN(x + attn) -> f32 + bf16 into concat[:, 0:256]
  add_layernorm<<<R, 256, 0, stream>>>(x, oproj, ln1g, ln1b, parf, catb, 512, 0);
  // recurrence projections: elu+1 feature map fused into the GEMM epilogue
  gemm_bf16_wmma<<<dim3(D / 64, R / 64), blk, 0, stream>>>(
      catb, 512, wrq, rq_b, nullptr, qm, D, R, D, D, 2);
  gemm_bf16_wmma<<<dim3(D / 64, R / 64), blk, 0, stream>>>(
      catb, 512, wrk, rk_b, nullptr, km, D, R, D, D, 2);
  gemm_bf16_wmma<<<dim3(D / 64, R / 64), blk, 0, stream>>>(
      catb, 512, wrv, rv_b, nullptr, vm, D, R, D, D, 0);
  // pack K / V heads for the recurrence WMMAs
  pack_head_bfrag<<<dim3(256, H, Bb), 256, 0, stream>>>(km, 256, 0, kpk, T);
  pack_head_bfrag<<<dim3(256, H, Bb), 256, 0, stream>>>(vm, 256, 0, vpk, T);
  // chunked linear recurrence -> bf16 [R,256]
  linrec_wmma<<<Bb * H, blk, 0, stream>>>(qm, km, kpk, vpk, seqb, T);
  // ro projection -> f32
  gemm_bf16_wmma<<<dim3(D / 64, R / 64), blk, 0, stream>>>(
      seqb, D, wro, ro_b, rop, nullptr, D, R, D, D, 0);
  // seq_repr = LN(parallel + seq) -> f32 + bf16 into concat[:, 256:512]
  add_layernorm<<<R, 256, 0, stream>>>(parf, rop, ln2g, ln2b, seqr, catb, 512, 256);
  // gate = sigmoid(concat @ cg_w + cg_b) -> f32 (sigmoid fused)
  gemm_bf16_wmma<<<dim3(D / 64, R / 64), blk, 0, stream>>>(
      catb, 512, wcg, cg_b, gatef, nullptr, D, R, D, 512, 3);
  // merged = g*parallel + (1-g)*seq_repr
  gate_merge<<<(unsigned)(((size_t)R * D + 255) / 256), 256, 0, stream>>>(
      gatef, parf, seqr, mergedf, mergedb, (size_t)R * D);
  // FFN: GELU fused into first GEMM; bf16 hidden
  gemm_bf16_wmma<<<dim3(HID / 64, R / 64), blk, 0, stream>>>(
      mergedb, D, w1b, b1, nullptr, ffnh, HID, R, HID, D, 1);
  gemm_bf16_wmma<<<dim3(D / 64, R / 64), blk, 0, stream>>>(
      ffnh, HID, w2b, b2, ffn2, nullptr, D, R, D, HID, 0);
  // final LN -> d_out (f32)
  add_layernorm<<<R, 256, 0, stream>>>(mergedf, ffn2, ln3g, ln3b, (float*)d_out,
                                       nullptr, 0, 0);
}
